// AudioQuantizer_56504589746624
// MI455X (gfx1250) — matrix operand order repaired and checked by reading, not compiled, not run on hardware
//
#include <hip/hip_runtime.h>

#define K_CODES 8192
#define D_DIM   512
#define N_ROWS  16384
#define M_TILE  128
#define KT      16                     // codebook rows per LDS tile
#define K_SPLIT 8
#define K_SLICE (K_CODES / K_SPLIT)    // 1024 codes per block slice

// LDS layout in bf16 elements, stride 512 with XOR-16 chunk swizzle (no padding):
//   XH  [128][512] @ 0
//   XL  [128][512] @ 65536
//   B   2 buffers x (hi [16][512], lo [16][512]) @ 131072
// Total = 163840 elems = 327680 bytes = exactly 320 KB.
#define XH_OFF 0
#define XL_OFF (M_TILE * D_DIM)
#define B_OFF  (2 * M_TILE * D_DIM)
#define B_BUF_ELEMS (2 * KT * D_DIM)   // hi+lo per buffer = 16384 elems

typedef __attribute__((ext_vector_type(16))) __bf16 v16bf;
typedef __attribute__((ext_vector_type(8)))  float  v8f;
typedef int v4i __attribute__((vector_size(16)));

union FragU { v16bf v; uint4 q[2]; };

__device__ __forceinline__ unsigned short f2bf(float f) {
    unsigned u = __float_as_uint(f);
    u += 0x7FFFu + ((u >> 16) & 1u);          // round-to-nearest-even
    return (unsigned short)(u >> 16);
}
__device__ __forceinline__ float bf2f(unsigned short h) {
    return __uint_as_float(((unsigned)h) << 16);
}
// swizzled element offset within a row: 8-elem (16B) chunk index XORed with row&15
__device__ __forceinline__ int swz(int row, int chunk) {
    return ((chunk ^ (row & 15)) << 3);
}

// ---- CDNA5 async global->LDS staging (guarded; falls back to sync path) ----
#if __has_builtin(__builtin_amdgcn_global_load_async_to_lds_b128)
#define HAVE_ASYNC_LDS 1
__device__ __forceinline__ void async_cp16(const unsigned short* g, unsigned short* l) {
    __builtin_amdgcn_global_load_async_to_lds_b128((v4i*)g, (v4i*)l, 0, 0);
}
__device__ __forceinline__ void wait_async0() {
#if __has_builtin(__builtin_amdgcn_s_wait_asynccnt)
    __builtin_amdgcn_s_wait_asynccnt(0);
#else
    asm volatile("s_wait_asynccnt 0x0" ::: "memory");
#endif
}
#endif

// ---------------------------------------------------------------------------
// Kernel 1: codebook fp32 -> bf16 hi/lo planes
// ---------------------------------------------------------------------------
__global__ void vq_conv_cb(const float* __restrict__ cb,
                           unsigned short* __restrict__ hi,
                           unsigned short* __restrict__ lo) {
    size_t i = ((size_t)blockIdx.x * 256 + threadIdx.x) * 4;
    float4 f = *(const float4*)(cb + i);
    unsigned short h0 = f2bf(f.x), h1 = f2bf(f.y), h2 = f2bf(f.z), h3 = f2bf(f.w);
    unsigned short l0 = f2bf(f.x - bf2f(h0));
    unsigned short l1 = f2bf(f.y - bf2f(h1));
    unsigned short l2 = f2bf(f.z - bf2f(h2));
    unsigned short l3 = f2bf(f.w - bf2f(h3));
    uint2 hp, lp;
    hp.x = (unsigned)h0 | ((unsigned)h1 << 16);
    hp.y = (unsigned)h2 | ((unsigned)h3 << 16);
    lp.x = (unsigned)l0 | ((unsigned)l1 << 16);
    lp.y = (unsigned)l2 | ((unsigned)l3 << 16);
    *(uint2*)(hi + i) = hp;
    *(uint2*)(lo + i) = lp;
}

// ---------------------------------------------------------------------------
// Kernel 2: half squared norms of codebook rows (one wave32 per row)
// ---------------------------------------------------------------------------
__global__ void vq_cnorm(const float* __restrict__ cb, float* __restrict__ cn) {
    int gwave = (int)((blockIdx.x * blockDim.x + threadIdx.x) >> 5);
    int lane  = threadIdx.x & 31;
    const float4* row = (const float4*)(cb + (size_t)gwave * D_DIM);
    float s = 0.f;
    for (int i = lane; i < D_DIM / 4; i += 32) {
        float4 v = row[i];
        s += v.x * v.x + v.y * v.y + v.z * v.z + v.w * v.w;
    }
    for (int m = 16; m >= 1; m >>= 1) s += __shfl_xor(s, m, 32);
    if (lane == 0) cn[gwave] = 0.5f * s;
}

// ---------------------------------------------------------------------------
// Kernel 3: WMMA GEMM (bf16 hi/lo, 3-term split) + fused running argmin
//           LDS double-buffered B tiles; async-to-LDS staging; 2x-unrolled
//           tile loop so all fragment addresses are loop-invariant
// ---------------------------------------------------------------------------
__global__ __launch_bounds__(256, 1)
void vq_main(const float* __restrict__ x,
             const unsigned short* __restrict__ cbh,
             const unsigned short* __restrict__ cbl,
             const float* __restrict__ cnorm,
             float* __restrict__ pbest,
             int* __restrict__ pidx) {
    extern __shared__ unsigned short lds[];
    unsigned short* XH = lds + XH_OFF;
    unsigned short* XL = lds + XL_OFF;

    const int tid  = threadIdx.x;
    const int lane = tid & 31;
    const int wave = tid >> 5;
    const int bk   = blockIdx.x & (K_SPLIT - 1);
    const int br   = blockIdx.x >> 3;
    const int row0 = br * M_TILE;
    const int k_begin = bk * K_SLICE;

    // ---- Stage A tile: fp32 x -> bf16 hi/lo directly into LDS (swizzled) ----
    {
        const float4* xg = (const float4*)(x + (size_t)row0 * D_DIM);
        for (int i = tid; i < (M_TILE * D_DIM) / 4; i += 256) {
            int r  = i >> 7;                  // 128 float4 per row
            int c4 = i & 127;
            float4 f = xg[(size_t)r * (D_DIM / 4) + c4];
            unsigned short h0 = f2bf(f.x), h1 = f2bf(f.y), h2 = f2bf(f.z), h3 = f2bf(f.w);
            unsigned short l0 = f2bf(f.x - bf2f(h0));
            unsigned short l1 = f2bf(f.y - bf2f(h1));
            unsigned short l2 = f2bf(f.z - bf2f(h2));
            unsigned short l3 = f2bf(f.w - bf2f(h3));
            uint2 hp, lp;
            hp.x = (unsigned)h0 | ((unsigned)h1 << 16);
            hp.y = (unsigned)h2 | ((unsigned)h3 << 16);
            lp.x = (unsigned)l0 | ((unsigned)l1 << 16);
            lp.y = (unsigned)l2 | ((unsigned)l3 << 16);
            int off = r * D_DIM + swz(r, c4 >> 1) + ((c4 & 1) << 2);   // 8B aligned
            *(uint2*)(XH + off) = hp;
            *(uint2*)(XL + off) = lp;
        }
    }

    float best[8];
    int   bidx[8];
#pragma unroll
    for (int r = 0; r < 8; ++r) { best[r] = 3.4e38f; bidx[r] = 0; }

    const int n = lane & 15;     // C/D column; B fragment row; == (arow & 15)
    const int h = lane >> 4;     // lane half -> K-offset select
    const int arow = wave * 16 + n;

    unsigned short* B0 = lds + B_OFF;
    unsigned short* B1 = B0 + B_BUF_ELEMS;

    // GEMM + argmin for one 16-code tile read from buffer base CH
    auto compute_tile = [&](const unsigned short* CH, int kbase) {
        const unsigned short* CL = CH + KT * D_DIM;
        v8f acc = {};
#pragma unroll
        for (int d0 = 0; d0 < D_DIM; d0 += 32) {
            const int ca0 = (d0 >> 3) + h;            // A chunk, K = d0+h*8..+7
            const int ca1 = ca0 + 2;                  // A chunk, K = d0+16+h*8..+7
            const int cb0 = (d0 >> 3) + h;            // B chunk, K = d0+h*8..+7
            const int cb1 = cb0 + 2;                  // B chunk, K = d0+16+h*8..+7
            FragU ah, al, bh, bl;
            const unsigned short* ahp = XH + arow * D_DIM;
            ah.q[0] = *(const uint4*)(ahp + swz(n, ca0));
            ah.q[1] = *(const uint4*)(ahp + swz(n, ca1));
            const unsigned short* alp = XL + arow * D_DIM;
            al.q[0] = *(const uint4*)(alp + swz(n, ca0));
            al.q[1] = *(const uint4*)(alp + swz(n, ca1));
            const unsigned short* bhp = CH + n * D_DIM;
            bh.q[0] = *(const uint4*)(bhp + swz(n, cb0));
            bh.q[1] = *(const uint4*)(bhp + swz(n, cb1));
            const unsigned short* blp = CL + n * D_DIM;
            bl.q[0] = *(const uint4*)(blp + swz(n, cb0));
            bl.q[1] = *(const uint4*)(blp + swz(n, cb1));

            acc = __builtin_amdgcn_wmma_f32_16x16x32_bf16(false, ah.v, false, bh.v,
                                                          (short)0, acc, false, false);
            acc = __builtin_amdgcn_wmma_f32_16x16x32_bf16(false, ah.v, false, bl.v,
                                                          (short)0, acc, false, false);
            acc = __builtin_amdgcn_wmma_f32_16x16x32_bf16(false, al.v, false, bh.v,
                                                          (short)0, acc, false, false);
        }
        const int   kabs = kbase + n;
        const float cn   = cnorm[kabs];               // score = 0.5||c||^2 - x.c
#pragma unroll
        for (int r = 0; r < 8; ++r) {
            float s = cn - acc[r];
            if (s < best[r]) { best[r] = s; bidx[r] = kabs; }
        }
    };

#if defined(HAVE_ASYNC_LDS)
    // ---- async staging: global -> LDS directly, no VGPR round trip ----
    auto asyncB = [&](int kabs0, unsigned short* dstH) {
        unsigned short* dstL = dstH + KT * D_DIM;
#pragma unroll
        for (int j = 0; j < 4; ++j) {
            int i  = tid + j * 256;           // 0..1023 chunk ids
            int r  = i >> 6;                  // 64 chunks per row
            int c8 = i & 63;
            size_t g = (size_t)(kabs0 + r) * D_DIM + (size_t)c8 * 8;
            int off = r * D_DIM + swz(r, c8);
            async_cp16(cbh + g, dstH + off);
            async_cp16(cbl + g, dstL + off);
        }
    };
    asyncB(k_begin, B0);
    wait_async0();
    __syncthreads();

#pragma unroll 1
    for (int k0 = 0; k0 < K_SLICE; k0 += 2 * KT) {
        // even tile: compute from B0 while async fills B1
        if (k0 + KT < K_SLICE) asyncB(k_begin + k0 + KT, B1);
        compute_tile(B0, k_begin + k0);
        wait_async0();
        __syncthreads();

        // odd tile: compute from B1 while async fills B0
        if (k0 + 2 * KT < K_SLICE) asyncB(k_begin + k0 + 2 * KT, B0);
        compute_tile(B1, k_begin + k0 + KT);
        wait_async0();
        __syncthreads();
    }
#else
    // ---- sync staging: global -> regs -> swizzled LDS ----
    uint4 ph[4], pl[4];
    auto prefetchB = [&](int kabs0) {
#pragma unroll
        for (int j = 0; j < 4; ++j) {
            int i  = tid + j * 256;
            int r  = i >> 6;
            int c8 = i & 63;
            size_t g = (size_t)(kabs0 + r) * D_DIM + (size_t)c8 * 8;
            ph[j] = *(const uint4*)(cbh + g);
            pl[j] = *(const uint4*)(cbl + g);
        }
    };
    auto storeB = [&](unsigned short* dstH) {
        unsigned short* dstL = dstH + KT * D_DIM;
#pragma unroll
        for (int j = 0; j < 4; ++j) {
            int i  = tid + j * 256;
            int r  = i >> 6;
            int c8 = i & 63;
            int off = r * D_DIM + swz(r, c8);
            *(uint4*)(dstH + off) = ph[j];
            *(uint4*)(dstL + off) = pl[j];
        }
    };
    prefetchB(k_begin);
    storeB(B0);
    __syncthreads();

#pragma unroll 1
    for (int k0 = 0; k0 < K_SLICE; k0 += 2 * KT) {
        if (k0 + KT < K_SLICE) prefetchB(k_begin + k0 + KT);
        compute_tile(B0, k_begin + k0);
        if (k0 + KT < K_SLICE) storeB(B1);
        __syncthreads();

        if (k0 + 2 * KT < K_SLICE) prefetchB(k_begin + k0 + 2 * KT);
        compute_tile(B1, k_begin + k0 + KT);
        if (k0 + 2 * KT < K_SLICE) storeB(B0);
        __syncthreads();
    }
#endif

    // reduce across the 16 lanes of each half (different N, same M set)
#pragma unroll
    for (int r = 0; r < 8; ++r) {
        float v = best[r];
        int   ix = bidx[r];
#pragma unroll
        for (int m = 1; m <= 8; m <<= 1) {
            float ov  = __shfl_xor(v, m, 32);
            int   oix = __shfl_xor(ix, m, 32);
            if (ov < v || (ov == v && oix < ix)) { v = ov; ix = oix; }
        }
        if (n == 0) {
            int row = row0 + wave * 16 + h * 8 + r;    // C row M = r + 8*h
            pbest[(size_t)row * K_SPLIT + bk] = v;
            pidx[(size_t)row * K_SPLIT + bk]  = ix;
        }
    }
}

// ---------------------------------------------------------------------------
// Kernel 4: merge K_SPLIT partial argmins per row + gather codebook row (fp32)
// ---------------------------------------------------------------------------
__global__ void vq_gather(const float* __restrict__ cb,
                          const float* __restrict__ pbest,
                          const int* __restrict__ pidx,
                          float* __restrict__ out) {
    int row = blockIdx.x * 2 + (threadIdx.x >> 7);
    int c   = threadIdx.x & 127;
    float bv = 3.4e38f;
    int   bix = 0;
#pragma unroll
    for (int s = 0; s < K_SPLIT; ++s) {
        float v  = pbest[(size_t)row * K_SPLIT + s];
        int   ix = pidx[(size_t)row * K_SPLIT + s];
        if (v < bv || (v == bv && ix < bix)) { bv = v; bix = ix; }
    }
    ((float4*)out)[(size_t)row * (D_DIM / 4) + c] =
        ((const float4*)cb)[(size_t)bix * (D_DIM / 4) + c];
}

// ---------------------------------------------------------------------------
extern "C" void kernel_launch(void* const* d_in, const int* in_sizes, int n_in,
                              void* d_out, int out_size, void* d_ws, size_t ws_size,
                              hipStream_t stream) {
    const float* x  = (const float*)d_in[0];   // [8,2048,512] fp32
    const float* cb = (const float*)d_in[1];   // [8192,512]  fp32
    float* out = (float*)d_out;

    char* ws = (char*)d_ws;
    unsigned short* cbh   = (unsigned short*)(ws);                      //  8 MB
    unsigned short* cbl   = (unsigned short*)(ws + 8388608);            //  8 MB
    float*          cnorm = (float*)(ws + 16777216);                    // 32 KB
    float*          pbest = (float*)(ws + 16777216 + 32768);            // 512 KB
    int*            pidx  = (int*)(ws + 16777216 + 32768 + 524288);     // 512 KB

    hipLaunchKernelGGL(vq_conv_cb, dim3((K_CODES * D_DIM) / 1024), dim3(256), 0, stream,
                       cb, cbh, cbl);
    hipLaunchKernelGGL(vq_cnorm, dim3(K_CODES / 8), dim3(256), 0, stream, cb, cnorm);

    size_t ldsBytes = (size_t)(B_OFF + 2 * B_BUF_ELEMS) * sizeof(unsigned short); // 327680
    (void)hipFuncSetAttribute(reinterpret_cast<const void*>(vq_main),
                              hipFuncAttributeMaxDynamicSharedMemorySize, (int)ldsBytes);
    hipLaunchKernelGGL(vq_main, dim3((N_ROWS / M_TILE) * K_SPLIT), dim3(256), ldsBytes, stream,
                       x, cbh, cbl, cnorm, pbest, pidx);

    hipLaunchKernelGGL(vq_gather, dim3(N_ROWS / 2), dim3(256), 0, stream,
                       cb, pbest, pidx, out);
}
